// GINModel_2645699854676
// MI455X (gfx1250) — matrix operand order repaired and checked by reading, not compiled, not run on hardware
//
#include <hip/hip_runtime.h>
#include <hip/hip_bf16.h>

// ---------------------------------------------------------------------------
// GIN model on MI455X (gfx1250):
//   h = emb[feats]
//   3x: z = h + scatter_add(h[src] -> dst);  h = relu(z@W1+b1)@W2 + b2  (WMMA)
//   out = segment_sum(h, graph_ids) @ Wr
//
// MLP computed transposed:  z2^T = W1^T @ z^T ; h^T = W2^T @ z2^T
//   -> weights are the A operand (pre-swizzled in LDS, 32B/lane fragments)
//   -> activations are the B operand (16 consecutive floats per lane)
//   -> inter-GEMM relayout = cross-half lane shuffle, no LDS round trip
// Producers dual-write h and z so the per-layer z=h copy kernel is gone.
// ---------------------------------------------------------------------------

#define N_NODES  50000
#define N_EDGES  800000
#define N_GRAPHS 128
#define HID      64

typedef __attribute__((ext_vector_type(16))) __bf16 v16bf;
typedef __attribute__((ext_vector_type(8)))  float  v8f;

// ---------------------------------------------------------------------------
// h[n][d] = z[n][d] = emb[feats[n]][d]    (16 threads per node, float4 each)
// ---------------------------------------------------------------------------
__global__ void embed_kernel(const int* __restrict__ feats,
                             const float* __restrict__ emb,
                             float* __restrict__ h,
                             float* __restrict__ z) {
    int t    = blockIdx.x * blockDim.x + threadIdx.x;
    int node = t >> 4;
    int c    = (t & 15) * 4;
    if (node >= N_NODES) return;
    const float4 v = *(const float4*)(emb + (size_t)feats[node] * HID + c);
    *(float4*)(h + (size_t)node * HID + c) = v;
    *(float4*)(z + (size_t)node * HID + c) = v;   // aggregation self term
}

// ---------------------------------------------------------------------------
// Edge scatter: z[dst[e]] += h[src[e]].  16 threads/edge, float4 per thread.
// 2 x 12.8MB working set is L2-resident (192MB), atomics run at L2 rate.
// ---------------------------------------------------------------------------
__global__ void scatter_kernel(const float* __restrict__ h,
                               float* __restrict__ z,
                               const int* __restrict__ src,
                               const int* __restrict__ dst) {
    long long t = (long long)blockIdx.x * blockDim.x + threadIdx.x;
    int e = (int)(t >> 4);
    int c = (int)(t & 15) * 4;
    if (e >= N_EDGES) return;
    const float4 v = *(const float4*)(h + (size_t)src[e] * HID + c);
    float* p = z + (size_t)dst[e] * HID + c;
    atomicAdd(p + 0, v.x);
    atomicAdd(p + 1, v.y);
    atomicAdd(p + 2, v.z);
    atomicAdd(p + 3, v.w);
}

// ---------------------------------------------------------------------------
// Fused MLP (transposed): 256 threads = 8 wave32s, 16 node-rows per wave.
// Reads z; writes h (and re-seeds z = h for the next layer when writeZ).
// ---------------------------------------------------------------------------
__global__ __launch_bounds__(256) void mlp_kernel(
    float* __restrict__ z, float* __restrict__ h,
    const float* __restrict__ W1, const float* __restrict__ b1,
    const float* __restrict__ W2, const float* __restrict__ b2,
    int writeZ) {

    // W^T in WMMA A-fragment order: Af[g][ks][mt][lane][e] (g=0:W1, 1:W2).
    // A tile (mt,ks): A[m][kl] = W[ks*32+kl][mt*16+m]; lane l16=m, half from kl.
    __shared__ __align__(32) __bf16 Af[2][2][4][32][16];   // 16 KB

    const int tid  = threadIdx.x;
    const int wave = tid >> 5;
    const int lane = tid & 31;
    const int half = lane >> 4;              // 0: lanes 0-15, 1: lanes 16-31
    const int l16  = lane & 15;

    const int row0 = (blockIdx.x * 8 + wave) * 16;   // wave's 16-node tile

    // ---- stage W1^T/W2^T as bf16 A-fragments into LDS --------------------
    for (int i = tid; i < HID * HID; i += 256) {
        const int k  = i >> 6, n = i & 63;           // W[k][n]
        const int ks = k >> 5, kl = k & 31;
        int hh, e;
        hh = kl >> 4; e = kl & 15;                                      // K=16h+e
        const int mt = n >> 4;
        const int fl = hh * 16 + (n & 15);
        Af[0][ks][mt][fl][e] = (__bf16)W1[i];
        Af[1][ks][mt][fl][e] = (__bf16)W2[i];
    }
    __syncthreads();

    auto loadA = [&](int g, int ks, int mt) -> v16bf {
        return *(const v16bf*)&Af[g][ks][mt][lane][0];   // 32B contiguous
    };

    // ---- B1 = z^T fragment: lane holds 16 consecutive feats of one row --
    int rowA = row0 + l16;
    if (rowA >= N_NODES) rowA = N_NODES - 1;             // clamp (stores guarded)
    const float* zr = z + (size_t)rowA * HID;

    auto loadBz = [&](int ks) -> v16bf {
        const float* p = zr + ks * 32 + half * 16;       // K = ks*32+16h+e
        v16bf b;
#pragma unroll
        for (int e = 0; e < 16; ++e) b[e] = (__bf16)p[e];
        return b;
    };

    // ---- GEMM1: z2^T[mt] = sum_ks W1^T[mt][ks] x z^T[ks] -----------------
    v8f acc[4] = {};
    {
        v16bf bz0 = loadBz(0);
        v16bf bz1 = loadBz(1);
#pragma unroll
        for (int mt = 0; mt < 4; ++mt) {
            acc[mt] = __builtin_amdgcn_wmma_f32_16x16x32_bf16(false, loadA(0, 0, mt), false, bz0, (short)0, acc[mt], false, false);
            acc[mt] = __builtin_amdgcn_wmma_f32_16x16x32_bf16(false, loadA(0, 1, mt), false, bz1, (short)0, acc[mt], false, false);
        }
    }

    // ---- bias + ReLU ------------------------------------------------------
    // D layout: acc[mt][r] = z2[node l16][feat mt*16 + 8*half + r]
    float t[4][8];
#pragma unroll
    for (int mt = 0; mt < 4; ++mt) {
        const float* bp = b1 + mt * 16 + 8 * half;
#pragma unroll
        for (int r = 0; r < 8; ++r)
            t[mt][r] = fmaxf(acc[mt][r] + bp[r], 0.0f);
    }

    // ---- B2 = z2^T fragment via cross-half lane exchange ------------------
    // Block q = ks*2 + half holds K = q*16 + (8*half_d + r); my half supplies
    // e in [8*half, 8*half+8), partner lane^16 supplies the other 8.
    auto makeB2 = [&](int ks) -> v16bf {
        v16bf b;
#pragma unroll
        for (int r = 0; r < 8; ++r) {
            float own  = half ? t[ks * 2 + 1][r] : t[ks * 2][r];     // my block q
            float send = half ? t[ks * 2][r]     : t[ks * 2 + 1][r]; // partner's q
            float recv = __shfl_xor(send, 16);
            b[r]     = (__bf16)(half ? recv : own);
            b[8 + r] = (__bf16)(half ? own  : recv);
        }
        return b;
    };

    // ---- GEMM2: h^T[mt] = sum_ks W2^T[mt][ks] x z2^T[ks] ------------------
    v8f acc2[4] = {};
    {
        v16bf b20 = makeB2(0);
        v16bf b21 = makeB2(1);
#pragma unroll
        for (int mt = 0; mt < 4; ++mt) {
            acc2[mt] = __builtin_amdgcn_wmma_f32_16x16x32_bf16(false, loadA(1, 0, mt), false, b20, (short)0, acc2[mt], false, false);
            acc2[mt] = __builtin_amdgcn_wmma_f32_16x16x32_bf16(false, loadA(1, 1, mt), false, b21, (short)0, acc2[mt], false, false);
        }
    }

    // ---- bias + store: lane owns 8 consecutive feats per mt -> b128 stores
    if (row0 + l16 < N_NODES) {
        float* hr = h + (size_t)(row0 + l16) * HID;
        float* zw = z + (size_t)(row0 + l16) * HID;     // next layer's self term
#pragma unroll
        for (int mt = 0; mt < 4; ++mt) {
            const float* bp = b2 + mt * 16 + 8 * half;
            float4 o0, o1;
            o0.x = acc2[mt][0] + bp[0]; o0.y = acc2[mt][1] + bp[1];
            o0.z = acc2[mt][2] + bp[2]; o0.w = acc2[mt][3] + bp[3];
            o1.x = acc2[mt][4] + bp[4]; o1.y = acc2[mt][5] + bp[5];
            o1.z = acc2[mt][6] + bp[6]; o1.w = acc2[mt][7] + bp[7];
            *(float4*)(hr + mt * 16 + 8 * half)     = o0;
            *(float4*)(hr + mt * 16 + 8 * half + 4) = o1;
            if (writeZ) {
                *(float4*)(zw + mt * 16 + 8 * half)     = o0;
                *(float4*)(zw + mt * 16 + 8 * half + 4) = o1;
            }
        }
    }
}

// ---------------------------------------------------------------------------
// Pooling + head
// ---------------------------------------------------------------------------
__global__ void zero_kernel(float* __restrict__ p, int n) {
    int t = blockIdx.x * blockDim.x + threadIdx.x;
    if (t < n) p[t] = 0.0f;
}

__global__ void pool_kernel(const float* __restrict__ h,
                            const int* __restrict__ gid,
                            float* __restrict__ pooled) {
    int t    = blockIdx.x * blockDim.x + threadIdx.x;
    int node = t >> 4;
    int c    = (t & 15) * 4;
    if (node >= N_NODES) return;
    int g = gid[node];
    const float4 v = *(const float4*)(h + (size_t)node * HID + c);
    float* p = pooled + (size_t)g * HID + c;
    atomicAdd(p + 0, v.x);
    atomicAdd(p + 1, v.y);
    atomicAdd(p + 2, v.z);
    atomicAdd(p + 3, v.w);
}

__global__ void head_kernel(const float* __restrict__ pooled,
                            const float* __restrict__ Wr,
                            float* __restrict__ out) {
    int g = blockIdx.x * blockDim.x + threadIdx.x;
    if (g >= N_GRAPHS) return;
    float s = 0.0f;
#pragma unroll
    for (int d = 0; d < HID; ++d) s += pooled[g * HID + d] * Wr[d];
    out[g] = s;
}

// ---------------------------------------------------------------------------
extern "C" void kernel_launch(void* const* d_in, const int* in_sizes, int n_in,
                              void* d_out, int out_size, void* d_ws, size_t ws_size,
                              hipStream_t stream) {
    const int*   feats = (const int*)d_in[0];
    const int*   src   = (const int*)d_in[1];
    const int*   dst   = (const int*)d_in[2];
    const int*   gid   = (const int*)d_in[3];
    const float* emb   = (const float*)d_in[4];
    const float* W1    = (const float*)d_in[5];   // [3][64][64]
    const float* b1    = (const float*)d_in[6];   // [3][64]
    const float* W2    = (const float*)d_in[7];
    const float* b2    = (const float*)d_in[8];
    const float* Wr    = (const float*)d_in[9];   // [64][1]
    float*       out   = (float*)d_out;           // [128]

    char* ws = (char*)d_ws;
    float* h      = (float*)(ws);                                   // 12.8 MB
    float* z      = (float*)(ws + (size_t)N_NODES * HID * 4);       // 12.8 MB
    float* pooled = (float*)(ws + (size_t)2 * N_NODES * HID * 4);   // 32 KB

    const int nodeVecThreads = N_NODES * 16;                 // 16 thr/node
    const int nodeVecBlocks  = (nodeVecThreads + 255) / 256; // 3125
    const long long edgeThreads = (long long)N_EDGES * 16;
    const int edgeBlocks     = (int)((edgeThreads + 255) / 256);    // 50000
    const int mlpBlocks      = (N_NODES + 127) / 128;               // 391 (8 waves x 16 rows)

    embed_kernel<<<nodeVecBlocks, 256, 0, stream>>>(feats, emb, h, z);

    for (int l = 0; l < 3; ++l) {
        scatter_kernel<<<edgeBlocks, 256, 0, stream>>>(h, z, src, dst);
        mlp_kernel<<<mlpBlocks, 256, 0, stream>>>(z, h,
                                                  W1 + (size_t)l * HID * HID,
                                                  b1 + (size_t)l * HID,
                                                  W2 + (size_t)l * HID * HID,
                                                  b2 + (size_t)l * HID,
                                                  /*writeZ=*/(l < 2) ? 1 : 0);
    }

    zero_kernel<<<(N_GRAPHS * HID + 255) / 256, 256, 0, stream>>>(pooled, N_GRAPHS * HID);
    pool_kernel<<<nodeVecBlocks, 256, 0, stream>>>(h, gid, pooled);
    head_kernel<<<1, 128, 0, stream>>>(pooled, Wr, out);
}
